// SelfAttention_1717986919024
// MI455X (gfx1250) — compile-verified, matrix operand order checked by reading, not run on hardware
//
#include <hip/hip_runtime.h>
#include <hip/hip_bf16.h>

// ---------------------------------------------------------------------------
// Self-attention forward for MI455X (gfx1250): bf16 WMMA everywhere.
//   q,k,v = x @ W{q,k,v}^T  (v_wmma_f32_16x16x32_bf16, fp32 accum)
//   rope+rmsnorm on q,k (1/sqrt(D) folded into q)
//   flash attention per (b,h), keys streamed in blocks of 32
//   y = attn_out @ Wproj^T
// Outputs: d_out = [ y (B*T*C fp32) | v1 (B*T*C fp32) ]
// ---------------------------------------------------------------------------

typedef __bf16 bf16_t;
typedef __attribute__((ext_vector_type(16))) __bf16 v16bf;
typedef __attribute__((ext_vector_type(8)))  __bf16 v8bf;
typedef __attribute__((ext_vector_type(8)))  float   v8f;
typedef __attribute__((ext_vector_type(4)))  unsigned int v4u;

#define NHEAD 16
#define DHEAD 128
#define CEMB  2048
#define TSEQ  2048
#define BATCH 4
#define MROWS (BATCH * TSEQ) /* 8192 */
#define EPSV  1.1920928955078125e-07f
#define QSCALE 0.08838834764831845f /* 1/sqrt(128) */

// Load a 16x32 bf16 A/B fragment: per ISA 7.12.2, lane holds row (lane&15),
// K-chunks at hi*8..hi*8+7 and 16+hi*8..16+hi*8+7 (two 16B loads).
static __device__ inline v16bf make_frag(const bf16_t* p0, const bf16_t* p1) {
  union { v16bf v; v4u u[2]; } f;
  f.u[0] = *(const v4u*)p0;
  f.u[1] = *(const v4u*)p1;
  return f.v;
}

static __device__ inline v8f wmma_bf16(v16bf a, v16bf b, v8f c) {
  return __builtin_amdgcn_wmma_f32_16x16x32_bf16(false, a, false, b,
                                                 (short)0, c, false, false);
}

// ---------------------------------------------------------------------------
// fp32 -> bf16 grid-stride convert
// ---------------------------------------------------------------------------
__global__ void f2bf_kernel(const float* __restrict__ in,
                            bf16_t* __restrict__ out, long n) {
  long i = (long)blockIdx.x * blockDim.x + threadIdx.x;
  long stride = (long)gridDim.x * blockDim.x;
  for (; i < n; i += stride) out[i] = (bf16_t)in[i];
}

// ---------------------------------------------------------------------------
// C = A[M,K] @ W[N,K]^T, bf16 in / fp32 accum.
// Wave computes 32(M) x 64(N): 2 A-frags + 4 B-frags -> 8 WMMAs per K=32
// (each B fragment reused for two M-tiles; 1.5 16B-loads per WMMA).
// mode 0: Cout[m*CEMB + n] = acc
// mode 1 (V): Cout gets v1 fp32; VT gets (l1+l2)*v as bf16 in [B,H,D,T]
// ---------------------------------------------------------------------------
__global__ __launch_bounds__(128)
void gemm_bf16_kernel(const bf16_t* __restrict__ A,
                      const bf16_t* __restrict__ W,
                      float* __restrict__ Cout,
                      bf16_t* __restrict__ VT,
                      const float* __restrict__ l1p,
                      const float* __restrict__ l2p,
                      int mode) {
  const int lane = threadIdx.x & 31;
  const int wave = threadIdx.x >> 5;
  const int lo = lane & 15, hi = lane >> 4;
  const int n0 = blockIdx.x * 64;
  const int m0 = blockIdx.y * 128 + wave * 32;

  v8f c[2][4];
#pragma unroll
  for (int mi = 0; mi < 2; ++mi)
#pragma unroll
    for (int j = 0; j < 4; ++j) c[mi][j] = {};

  const bf16_t* arow0 = A + (size_t)(m0 + lo) * CEMB;
  const bf16_t* arow1 = arow0 + (size_t)16 * CEMB;

  for (int k = 0; k < CEMB; k += 32) {
    // prefetch A one K-group ahead (lowers to global_prefetch_b8)
    if (k + 32 < CEMB) {
      __builtin_prefetch(arow0 + k + 32 + hi * 8, 0, 1);
      __builtin_prefetch(arow1 + k + 32 + hi * 8, 0, 1);
    }
    v16bf a0 = make_frag(arow0 + k + hi * 8, arow0 + k + 16 + hi * 8);
    v16bf a1 = make_frag(arow1 + k + hi * 8, arow1 + k + 16 + hi * 8);
#pragma unroll
    for (int j = 0; j < 4; ++j) {
      const bf16_t* brow = W + (size_t)(n0 + j * 16 + lo) * CEMB;
      v16bf bf = make_frag(brow + k + hi * 8, brow + k + 16 + hi * 8);
      c[0][j] = wmma_bf16(a0, bf, c[0][j]);
      c[1][j] = wmma_bf16(a1, bf, c[1][j]);
    }
  }

  // C/D layout: lane holds column (lane&15), rows hi*8 + vgpr (7.12.2).
  if (mode == 0) {
#pragma unroll
    for (int mi = 0; mi < 2; ++mi) {
      const int mbase = m0 + mi * 16 + hi * 8;
#pragma unroll
      for (int j = 0; j < 4; ++j) {
        int n = n0 + j * 16 + lo;
#pragma unroll
        for (int r = 0; r < 8; ++r)
          Cout[(size_t)(mbase + r) * CEMB + n] = c[mi][j][r];
      }
    }
  } else {
    const float ls = l1p[0] + l2p[0];
#pragma unroll
    for (int mi = 0; mi < 2; ++mi) {
      const int m = m0 + mi * 16 + hi * 8;
      const int b = m >> 11;            // /TSEQ
      const int t = m & (TSEQ - 1);
#pragma unroll
      for (int j = 0; j < 4; ++j) {
        int n = n0 + j * 16 + lo;
        int h = n >> 7, d = n & (DHEAD - 1);
        v8bf pk;
#pragma unroll
        for (int r = 0; r < 8; ++r) {
          float v = c[mi][j][r];
          Cout[(size_t)(m + r) * CEMB + n] = v;  // v1 (fp32)
          pk[r] = (bf16_t)(ls * v);
        }
        // [B,H,D,T]: 8 consecutive t -> one 16B store
        *(v8bf*)(VT + (((size_t)(b * NHEAD + h) * DHEAD + d) * TSEQ + t)) = pk;
      }
    }
  }
}

// ---------------------------------------------------------------------------
// RoPE + RMSNorm. One wave per (b,t,h); lane handles rotation pairs
// (lane, lane+64) and (lane+32, lane+96). 1/sqrt(D) folded into q.
// Output layout [B,H,T,D] bf16.
// ---------------------------------------------------------------------------
static __device__ inline void rope_one(const float* __restrict__ src,
                                       bf16_t* __restrict__ dst,
                                       const float* cp, const float* sp,
                                       long bt, int b, int t, int h,
                                       int lane, float outscale) {
  const float* x = src + (size_t)bt * CEMB + h * DHEAD;
  int j0 = lane, j1 = lane + 32;
  float x1a = x[j0], x2a = x[j0 + 64];
  float x1b = x[j1], x2b = x[j1 + 64];
  float ca = cp[j0], sa = sp[j0], cb = cp[j1], sb = sp[j1];
  float o0 =  x1a * ca + x2a * sa;
  float o2 = -x1a * sa + x2a * ca;
  float o1 =  x1b * cb + x2b * sb;
  float o3 = -x1b * sb + x2b * cb;
  float ss = o0 * o0 + o1 * o1 + o2 * o2 + o3 * o3;
#pragma unroll
  for (int m = 16; m; m >>= 1) ss += __shfl_xor(ss, m, 32);
  float rn = rsqrtf(ss * (1.0f / 128.0f) + EPSV) * outscale;
  bf16_t* o = dst + (((size_t)b * NHEAD + h) * TSEQ + t) * DHEAD;
  o[j0] = (bf16_t)(o0 * rn);
  o[j1] = (bf16_t)(o1 * rn);
  o[j0 + 64] = (bf16_t)(o2 * rn);
  o[j1 + 64] = (bf16_t)(o3 * rn);
}

__global__ __launch_bounds__(128)
void rope_rms_kernel(const float* __restrict__ Q, const float* __restrict__ K,
                     const float* __restrict__ cosp,
                     const float* __restrict__ sinp,
                     bf16_t* __restrict__ QH, bf16_t* __restrict__ KH) {
  const int wave = threadIdx.x >> 5, lane = threadIdx.x & 31;
  long idx = (long)blockIdx.x * 4 + wave;      // over B*T*H
  int h = (int)(idx % NHEAD);
  long bt = idx / NHEAD;
  int t = (int)(bt & (TSEQ - 1));
  int b = (int)(bt >> 11);
  const float* cp = cosp + (size_t)t * 64;
  const float* sp = sinp + (size_t)t * 64;
  rope_one(Q, QH, cp, sp, bt, b, t, h, lane, QSCALE);
  rope_one(K, KH, cp, sp, bt, b, t, h, lane, 1.0f);
}

// ---------------------------------------------------------------------------
// Flash attention. Wave owns 16 q-rows of one (b,h); streams keys in blocks
// of 32. QK^T: 8 WMMAs; P re-layout through LDS; P@V: 8 WMMAs into 16x128
// fp32 accumulator. Online softmax with per-lane row stats.
// ---------------------------------------------------------------------------
__global__ __launch_bounds__(128)
void attention_kernel(const bf16_t* __restrict__ QH,
                      const bf16_t* __restrict__ KH,
                      const bf16_t* __restrict__ VT,
                      bf16_t* __restrict__ YB) {
  __shared__ bf16_t pscr[4][16][32];
  const int wave = threadIdx.x >> 5, lane = threadIdx.x & 31;
  const int lo = lane & 15, hi = lane >> 4;
  const int nqc = TSEQ / 64;
  const int bh = blockIdx.x / nqc;
  const int q0 = (blockIdx.x % nqc) * 64 + wave * 16;
  const int b = bh >> 4, h = bh & 15;

  const bf16_t* qbase = QH + (size_t)bh * TSEQ * DHEAD;
  const bf16_t* kbase = KH + (size_t)bh * TSEQ * DHEAD;
  const bf16_t* vbase = VT + (size_t)bh * DHEAD * TSEQ;

  // Q fragments (16 rows x full D=128), loaded once
  v16bf qa[4];
  const bf16_t* qrow = qbase + (size_t)(q0 + lo) * DHEAD;
#pragma unroll
  for (int dk = 0; dk < 4; ++dk)
    qa[dk] = make_frag(qrow + dk * 32 + hi * 8, qrow + dk * 32 + 16 + hi * 8);

  v8f o[8];
#pragma unroll
  for (int j = 0; j < 8; ++j) o[j] = {};
  float mrow[8], lrow[8];
#pragma unroll
  for (int r = 0; r < 8; ++r) { mrow[r] = -1e30f; lrow[r] = 0.0f; }

  for (int kb = 0; kb < TSEQ; kb += 32) {
    // ---- S = q^ @ k^^T (scale pre-folded into q^) ----
    v8f s0 = {}, s1 = {};
    const bf16_t* k0row = kbase + (size_t)(kb + lo) * DHEAD;
    const bf16_t* k1row = k0row + 16 * DHEAD;
#pragma unroll
    for (int dk = 0; dk < 4; ++dk) {
      v16bf b0 = make_frag(k0row + dk * 32 + hi * 8, k0row + dk * 32 + 16 + hi * 8);
      s0 = wmma_bf16(qa[dk], b0, s0);
      v16bf b1 = make_frag(k1row + dk * 32 + hi * 8, k1row + dk * 32 + 16 + hi * 8);
      s1 = wmma_bf16(qa[dk], b1, s1);
    }

    // ---- online softmax (row = hi*8 + r, reduce across the 16-lane half) --
    float alpha[8];
#pragma unroll
    for (int r = 0; r < 8; ++r) {
      float mx = fmaxf(s0[r], s1[r]);
#pragma unroll
      for (int msk = 8; msk; msk >>= 1) mx = fmaxf(mx, __shfl_xor(mx, msk, 16));
      float mnew = fmaxf(mrow[r], mx);
      alpha[r] = __expf(mrow[r] - mnew);
      mrow[r] = mnew;
      float p0 = __expf(s0[r] - mnew);
      float p1 = __expf(s1[r] - mnew);
      s0[r] = p0; s1[r] = p1;
      float rs = p0 + p1;
#pragma unroll
      for (int msk = 8; msk; msk >>= 1) rs += __shfl_xor(rs, msk, 16);
      lrow[r] = lrow[r] * alpha[r] + rs;
    }
#pragma unroll
    for (int j = 0; j < 8; ++j)
#pragma unroll
      for (int r = 0; r < 8; ++r) o[j][r] *= alpha[r];

    // ---- P: C-fragment layout -> bf16 A-fragment via LDS bounce ----
    __syncthreads();
#pragma unroll
    for (int r = 0; r < 8; ++r) {
      pscr[wave][hi * 8 + r][lo]      = (bf16_t)s0[r];
      pscr[wave][hi * 8 + r][16 + lo] = (bf16_t)s1[r];
    }
    __syncthreads();
    v16bf pa = make_frag(&pscr[wave][lo][hi * 8], &pscr[wave][lo][16 + hi * 8]);

    // ---- O += P @ V  (V stored [D,T] so B-frag rows are d, K is t) ----
#pragma unroll
    for (int j = 0; j < 8; ++j) {
      const bf16_t* vrow = vbase + (size_t)(j * 16 + lo) * TSEQ + kb;
      v16bf bv = make_frag(vrow + hi * 8, vrow + 16 + hi * 8);
      o[j] = wmma_bf16(pa, bv, o[j]);
    }
  }

  // ---- finalize: divide by l, store bf16 into [B*T, C] ----
  float rl[8];
#pragma unroll
  for (int r = 0; r < 8; ++r) rl[r] = 1.0f / lrow[r];
  bf16_t* yrow = YB + ((size_t)(b * TSEQ + q0 + hi * 8)) * CEMB + h * DHEAD;
#pragma unroll
  for (int j = 0; j < 8; ++j)
#pragma unroll
    for (int r = 0; r < 8; ++r)
      yrow[(size_t)r * CEMB + j * 16 + lo] = (bf16_t)(o[j][r] * rl[r]);
}

// ---------------------------------------------------------------------------
extern "C" void kernel_launch(void* const* d_in, const int* in_sizes, int n_in,
                              void* d_out, int out_size, void* d_ws,
                              size_t ws_size, hipStream_t stream) {
  (void)in_sizes; (void)n_in; (void)out_size; (void)ws_size;
  const float* x     = (const float*)d_in[0];
  const float* wq    = (const float*)d_in[1];
  const float* wk    = (const float*)d_in[2];
  const float* wv    = (const float*)d_in[3];
  const float* wproj = (const float*)d_in[4];
  const float* lamb1 = (const float*)d_in[5];
  const float* lamb2 = (const float*)d_in[6];
  const float* cosp  = (const float*)d_in[7];
  const float* sinp  = (const float*)d_in[8];
  float* out = (float*)d_out;
  float* v1out = out + (size_t)MROWS * CEMB;

  char* wsb = (char*)d_ws;
  size_t off = 0;
  auto take = [&](size_t bytes) -> void* {
    off = (off + 255) & ~(size_t)255;
    void* p = wsb + off;
    off += bytes;
    return p;
  };
  bf16_t* xb  = (bf16_t*)take((size_t)MROWS * CEMB * 2);
  bf16_t* wqb = (bf16_t*)take((size_t)CEMB * CEMB * 2);
  bf16_t* wkb = (bf16_t*)take((size_t)CEMB * CEMB * 2);
  bf16_t* wvb = (bf16_t*)take((size_t)CEMB * CEMB * 2);
  bf16_t* wpb = (bf16_t*)take((size_t)CEMB * CEMB * 2);
  float*  qf  = (float*)take((size_t)MROWS * CEMB * 4);
  float*  kf  = (float*)take((size_t)MROWS * CEMB * 4);
  bf16_t* qh  = (bf16_t*)take((size_t)MROWS * CEMB * 2);
  bf16_t* kh  = (bf16_t*)take((size_t)MROWS * CEMB * 2);
  bf16_t* vt  = (bf16_t*)take((size_t)MROWS * CEMB * 2);
  bf16_t* yb  = (bf16_t*)take((size_t)MROWS * CEMB * 2);

  // 1) fp32 -> bf16 conversions
  f2bf_kernel<<<8192, 256, 0, stream>>>(x, xb, (long)MROWS * CEMB);
  f2bf_kernel<<<2048, 256, 0, stream>>>(wq, wqb, (long)CEMB * CEMB);
  f2bf_kernel<<<2048, 256, 0, stream>>>(wk, wkb, (long)CEMB * CEMB);
  f2bf_kernel<<<2048, 256, 0, stream>>>(wv, wvb, (long)CEMB * CEMB);
  f2bf_kernel<<<2048, 256, 0, stream>>>(wproj, wpb, (long)CEMB * CEMB);

  // 2) QKV GEMMs  (wave tile 32x64 -> grid 32 x 64)
  dim3 gg(CEMB / 64, MROWS / 128);
  gemm_bf16_kernel<<<gg, 128, 0, stream>>>(xb, wqb, qf, nullptr, lamb1, lamb2, 0);
  gemm_bf16_kernel<<<gg, 128, 0, stream>>>(xb, wkb, kf, nullptr, lamb1, lamb2, 0);
  gemm_bf16_kernel<<<gg, 128, 0, stream>>>(xb, wvb, v1out, vt, lamb1, lamb2, 1);

  // 3) RoPE + RMSNorm on q,k
  rope_rms_kernel<<<(MROWS * NHEAD) / 4, 128, 0, stream>>>(qf, kf, cosp, sinp,
                                                           qh, kh);

  // 4) Flash attention
  attention_kernel<<<BATCH * NHEAD * (TSEQ / 64), 128, 0, stream>>>(qh, kh, vt,
                                                                    yb);

  // 5) Output projection
  gemm_bf16_kernel<<<gg, 128, 0, stream>>>(yb, wpb, out, nullptr, lamb1, lamb2,
                                           0);
}
